// FeatureDecodingModule_14027363188878
// MI455X (gfx1250) — matrix operand order repaired and checked by reading, not compiled, not run on hardware
//
#include <hip/hip_runtime.h>
#include <hip/hip_bf16.h>
#include <stdint.h>

// ---------------------------------------------------------------------------
// FeatureDecodingModule for MI455X (gfx1250, wave32, WMMA).
//
// Shapes: B=8, N=8192, S=2048, D1=128, D2=256, IN=384, C0=C1=256.
// GEMM path: f16 A/B, f32 accumulate via v_wmma_f32_16x16x32_f16.
// Activations pre-packed into exact WMMA B-fragment order; each wave computes
// a 64(M)x16(N) C stripe so every B fragment load feeds 4 WMMAs.
// ---------------------------------------------------------------------------

typedef __attribute__((ext_vector_type(16))) _Float16 v16h;
typedef __attribute__((ext_vector_type(8)))  float    v8f;

static constexpr int B_    = 8;
static constexpr int N_    = 8192;
static constexpr int S_    = 2048;
static constexpr int D1_   = 128;
static constexpr int D2_   = 256;
static constexpr int IN_   = 384;        // D1 + D2
static constexpr int C_    = 256;        // C0 == C1
static constexpr int NCOL  = B_ * N_;    // 65536 GEMM columns
static constexpr int CT_   = NCOL / 16;  // 4096 column tiles
static constexpr int KB0_  = IN_ / 32;   // 12 K-blocks for conv0
static constexpr int KB1_  = C_  / 32;   // 8  K-blocks for conv1
static constexpr int MT_   = C_ / 16;    // 16 M tiles
static constexpr float BN_EPS = 1e-5f;

// ---------------------------------------------------------------------------
// Weight packing: A-fragment (16x32 f16) lane/half layout used here (contiguous K, same order as B):
//   lane l holds row M = l%16
//   lanes  0-15: half h -> K = h        (K 0-15)
//   lanes 16-31: half h -> K = 16 + h   (K 16-31)
// ---------------------------------------------------------------------------
__global__ void pack_w_kernel(const float* __restrict__ W, _Float16* __restrict__ Wp,
                              int KB, int K) {
    int idx = blockIdx.x * blockDim.x + threadIdx.x;
    int total = MT_ * KB * 512;
    if (idx >= total) return;
    int h    = idx & 15;
    int lane = (idx >> 4) & 31;
    int frag = idx >> 9;
    int kb    = frag % KB;
    int mtile = frag / KB;
    int M = mtile * 16 + (lane & 15);
    int Kk = kb * 32 + ((lane >> 4) << 4) + h;
    Wp[idx] = (_Float16)W[M * K + Kk];
}

// ---------------------------------------------------------------------------
// B-fragment (32x16 f16) layout: lane l -> column = l%16,
//   lanes 0-15 hold K 0-15 (h -> K=h), lanes 16-31 hold K 16-31 (h -> K=16+h).
// Packed buffer: halves[ ((kb*CT_ + ct)*32 + lane)*16 + h ].
// ---------------------------------------------------------------------------
__device__ __forceinline__ void write_b_frag(_Float16* __restrict__ Xp,
                                             int k, int col, float v) {
    int kb = k >> 5, kk = k & 31;
    int ct = col >> 4;
    int lane = ((kk >> 4) << 4) | (col & 15);
    int h = kk & 15;
    Xp[((size_t)(kb * CT_ + ct) * 32 + lane) * 16 + h] = (_Float16)v;
}

// ---------------------------------------------------------------------------
// 3-NN interpolation + concat. One thread per query point; xyz2 staged in LDS
// (3*2048 floats = 24KB, uniform index -> LDS broadcast reads).
// ---------------------------------------------------------------------------
__global__ void interp_concat_kernel(const float* __restrict__ xyz1,
                                     const float* __restrict__ xyz2,
                                     const float* __restrict__ points1,
                                     const float* __restrict__ points2,
                                     _Float16* __restrict__ Xp) {
    __shared__ float sx[3 * S_];
    const int tid = threadIdx.x;
    const int b   = blockIdx.y;
    const int n   = blockIdx.x * 256 + tid;

    const float* x2b = xyz2 + (size_t)b * 3 * S_;
    #pragma unroll
    for (int i = 0; i < (3 * S_) / 256; ++i)
        sx[tid + i * 256] = x2b[tid + i * 256];
    __syncthreads();

    const float x = xyz1[(size_t)b * 3 * N_ + 0 * N_ + n];
    const float y = xyz1[(size_t)b * 3 * N_ + 1 * N_ + n];
    const float z = xyz1[(size_t)b * 3 * N_ + 2 * N_ + n];

    float d0 = 3.4e38f, d1 = 3.4e38f, d2 = 3.4e38f;
    int   i0 = 0, i1 = 0, i2 = 0;
    #pragma unroll 4
    for (int s = 0; s < S_; ++s) {
        float dx = x - sx[s];
        float dy = y - sx[S_ + s];
        float dz = z - sx[2 * S_ + s];
        float d  = fmaf(dx, dx, fmaf(dy, dy, dz * dz));
        if (d < d2) {
            if (d < d1) {
                if (d < d0) { d2 = d1; i2 = i1; d1 = d0; i1 = i0; d0 = d; i0 = s; }
                else        { d2 = d1; i2 = i1; d1 = d;  i1 = s; }
            } else          { d2 = d;  i2 = s; }
        }
    }
    float r0 = 1.0f / (d0 + 1e-8f);
    float r1 = 1.0f / (d1 + 1e-8f);
    float r2 = 1.0f / (d2 + 1e-8f);
    float inv = 1.0f / (r0 + r1 + r2);
    float w0 = r0 * inv, w1 = r1 * inv, w2 = r2 * inv;

    const int col = b * N_ + n;
    const float* p1b = points1 + (size_t)b * D1_ * N_;
    #pragma unroll 4
    for (int k = 0; k < D1_; ++k)
        write_b_frag(Xp, k, col, p1b[(size_t)k * N_ + n]);

    const float* p2b = points2 + (size_t)b * D2_ * S_;
    #pragma unroll 4
    for (int c = 0; c < D2_; ++c) {
        const float* row = p2b + (size_t)c * S_;
        float v = w0 * row[i0] + w1 * row[i1] + w2 * row[i2];
        write_b_frag(Xp, D1_ + c, col, v);
    }
}

// ---------------------------------------------------------------------------
// GEMM: Y[c, col] = sum_k Wp[c,k] * Xp[k,col] + bias[c].
// One wave -> 64(M) x 16(N) C stripe: per K-block one B-fragment load feeds
// 4 WMMAs (4x activation reuse). All 8 waves of a block share one M-group so
// the block's weight working set is 4*KB KB-sized fragments (L2/WGP$-hot).
// EXEC is all-ones (full blocks, no divergence) as WMMA requires.
// ---------------------------------------------------------------------------
template <int KB>
__global__ void gemm_wmma_kernel(const _Float16* __restrict__ Wp,
                                 const _Float16* __restrict__ Xp,
                                 const float* __restrict__ bias,
                                 float* __restrict__ Y) {
    const int lane = threadIdx.x & 31;
    const int wave = threadIdx.x >> 5;
    const int ct   = blockIdx.x * 8 + wave;   // column tile
    const int mg   = blockIdx.y;              // M group: mtiles mg*4 .. mg*4+3

    const _Float16* bp = Xp + ((size_t)ct * 32 + lane) * 16;
    const _Float16* ap = Wp + ((size_t)(mg * 4) * KB * 32 + lane) * 16;

    v8f acc[4] = {{}, {}, {}, {}};
    #pragma unroll
    for (int kb = 0; kb < KB; ++kb) {
        v16h b = *reinterpret_cast<const v16h*>(bp + (size_t)kb * CT_ * 512);
        if (kb + 1 < KB)  // gfx1250 global_prefetch_b8 on next B fragment
            __builtin_prefetch(bp + (size_t)(kb + 1) * CT_ * 512, 0, 0);
        #pragma unroll
        for (int m = 0; m < 4; ++m) {
            v16h a = *reinterpret_cast<const v16h*>(ap + (size_t)(m * KB + kb) * 512);
            acc[m] = __builtin_amdgcn_wmma_f32_16x16x32_f16(
                         false, a, false, b, (short)0, acc[m], false, false);
        }
    }

    const int col = ct * 16 + (lane & 15);
    #pragma unroll
    for (int m = 0; m < 4; ++m) {
        const int rowB = (mg * 4 + m) * 16 + ((lane >> 4) << 3);
        #pragma unroll
        for (int r = 0; r < 8; ++r) {
            int c = rowB + r;
            Y[(size_t)c * NCOL + col] = acc[m][r] + bias[c];
        }
    }
}

// ---------------------------------------------------------------------------
// Per-channel BN stats: one block per channel, double accumulators.
// ---------------------------------------------------------------------------
__global__ void bn_stats_kernel(const float* __restrict__ Y,
                                float* __restrict__ meanArr,
                                float* __restrict__ rstdArr) {
    __shared__ double ss[256], ss2[256];
    const int c = blockIdx.x, tid = threadIdx.x;
    double s = 0.0, s2 = 0.0;
    const float* row = Y + (size_t)c * NCOL;
    for (int col = tid; col < NCOL; col += 256) {
        double v = (double)row[col];
        s += v; s2 += v * v;
    }
    ss[tid] = s; ss2[tid] = s2;
    __syncthreads();
    for (int off = 128; off > 0; off >>= 1) {
        if (tid < off) { ss[tid] += ss[tid + off]; ss2[tid] += ss2[tid + off]; }
        __syncthreads();
    }
    if (tid == 0) {
        double m   = ss[0]  / (double)NCOL;
        double var = ss2[0] / (double)NCOL - m * m;
        meanArr[c] = (float)m;
        rstdArr[c] = rsqrtf((float)var + BN_EPS);
    }
}

// ---------------------------------------------------------------------------
// BN + ReLU + repack into B-fragment f16 via LDS tile (coalesced both sides).
// grid = (NCOL/128, KB1_); block covers a 32(channel) x 128(col) tile.
// ---------------------------------------------------------------------------
__global__ void bn_relu_pack_kernel(const float* __restrict__ Y,
                                    const float* __restrict__ g,
                                    const float* __restrict__ beta,
                                    const float* __restrict__ meanArr,
                                    const float* __restrict__ rstdArr,
                                    _Float16* __restrict__ Xp) {
    __shared__ _Float16 tile[32 * 128];
    const int tid = threadIdx.x;
    const int cb  = blockIdx.x;   // 128-column block
    const int kb  = blockIdx.y;   // 32-channel block

    #pragma unroll
    for (int i = 0; i < 16; ++i) {
        int e = tid + i * 256;
        int cl = e >> 7, coll = e & 127;
        int c = kb * 32 + cl;
        int col = cb * 128 + coll;
        float v = (Y[(size_t)c * NCOL + col] - meanArr[c]) * rstdArr[c] * g[c] + beta[c];
        tile[cl * 128 + coll] = (_Float16)fmaxf(v, 0.0f);
    }
    __syncthreads();

    uint32_t* out32 = reinterpret_cast<uint32_t*>(Xp);
    const int lane  = tid >> 3;          // q0 = 2*tid -> lane = q0/16
    const int h0    = (tid << 1) & 15;
    const int c0    = ((lane >> 4) << 4) + h0;   // local channel of low half
    const int col16 = lane & 15;
    #pragma unroll
    for (int sub = 0; sub < 8; ++sub) {
        int ct   = cb * 8 + sub;
        int coll = sub * 16 + col16;
        union { _Float16 h[2]; uint32_t u; } p;
        p.h[0] = tile[c0 * 128 + coll];
        p.h[1] = tile[(c0 + 1) * 128 + coll];
        out32[(size_t)(kb * CT_ + ct) * 256 + tid] = p.u;
    }
}

// ---------------------------------------------------------------------------
// Final BN + ReLU -> output [B, C, N] fp32.
// ---------------------------------------------------------------------------
__global__ void bn_relu_out_kernel(const float* __restrict__ Y,
                                   const float* __restrict__ g,
                                   const float* __restrict__ beta,
                                   const float* __restrict__ meanArr,
                                   const float* __restrict__ rstdArr,
                                   float* __restrict__ out) {
    size_t idx = (size_t)blockIdx.x * 256 + threadIdx.x;  // c*NCOL + col
    int c   = (int)(idx >> 16);          // NCOL = 65536
    int col = (int)(idx & (NCOL - 1));
    float v = (Y[idx] - meanArr[c]) * rstdArr[c] * g[c] + beta[c];
    v = fmaxf(v, 0.0f);
    int b = col >> 13, n = col & (N_ - 1);
    out[((size_t)b * C_ + c) * N_ + n] = v;
}

// ---------------------------------------------------------------------------
extern "C" void kernel_launch(void* const* d_in, const int* in_sizes, int n_in,
                              void* d_out, int out_size, void* d_ws, size_t ws_size,
                              hipStream_t stream) {
    const float* xyz1    = (const float*)d_in[0];
    const float* xyz2    = (const float*)d_in[1];
    const float* points1 = (const float*)d_in[2];
    const float* points2 = (const float*)d_in[3];
    const float* conv_w0 = (const float*)d_in[4];
    const float* conv_b0 = (const float*)d_in[5];
    const float* bn_g0   = (const float*)d_in[6];
    const float* bn_b0   = (const float*)d_in[7];
    const float* conv_w1 = (const float*)d_in[8];
    const float* conv_b1 = (const float*)d_in[9];
    const float* bn_g1   = (const float*)d_in[10];
    const float* bn_b1   = (const float*)d_in[11];
    float* out = (float*)d_out;

    // Workspace carve-up (~118 MB total).
    char* ws = (char*)d_ws;
    _Float16* Xp  = (_Float16*)ws;                         // 50.3 MB (reused as X1p)
    size_t off = (size_t)KB0_ * CT_ * 512 * sizeof(_Float16);
    float* Ybuf   = (float*)(ws + off);                    // 67.1 MB
    off += (size_t)C_ * NCOL * sizeof(float);
    _Float16* W0p = (_Float16*)(ws + off);                 // 192 KB
    off += (size_t)MT_ * KB0_ * 512 * sizeof(_Float16);
    _Float16* W1p = (_Float16*)(ws + off);                 // 128 KB
    off += (size_t)MT_ * KB1_ * 512 * sizeof(_Float16);
    float* mean0 = (float*)(ws + off); off += C_ * sizeof(float);
    float* rstd0 = (float*)(ws + off); off += C_ * sizeof(float);
    float* mean1 = (float*)(ws + off); off += C_ * sizeof(float);
    float* rstd1 = (float*)(ws + off); off += C_ * sizeof(float);

    // 1) Pack weights into WMMA A-fragment order.
    pack_w_kernel<<<(MT_ * KB0_ * 512 + 255) / 256, 256, 0, stream>>>(conv_w0, W0p, KB0_, IN_);
    pack_w_kernel<<<(MT_ * KB1_ * 512 + 255) / 256, 256, 0, stream>>>(conv_w1, W1p, KB1_, C_);

    // 2) 3-NN interpolation + concat -> packed f16 B fragments.
    interp_concat_kernel<<<dim3(N_ / 256, B_), 256, 0, stream>>>(xyz1, xyz2, points1, points2, Xp);

    // 3) Conv0 GEMM (256x384 @ 384x65536) + bias.
    gemm_wmma_kernel<KB0_><<<dim3(CT_ / 8, MT_ / 4), 256, 0, stream>>>(W0p, Xp, conv_b0, Ybuf);

    // 4) BN0 stats.
    bn_stats_kernel<<<C_, 256, 0, stream>>>(Ybuf, mean0, rstd0);

    // 5) BN0 + ReLU + repack as f16 B fragments (into Xp region, now dead).
    bn_relu_pack_kernel<<<dim3(NCOL / 128, KB1_), 256, 0, stream>>>(Ybuf, bn_g0, bn_b0, mean0, rstd0, Xp);

    // 6) Conv1 GEMM (256x256 @ 256x65536) + bias.
    gemm_wmma_kernel<KB1_><<<dim3(CT_ / 8, MT_ / 4), 256, 0, stream>>>(W1p, Xp, conv_b1, Ybuf);

    // 7) BN1 stats.
    bn_stats_kernel<<<C_, 256, 0, stream>>>(Ybuf, mean1, rstd1);

    // 8) BN1 + ReLU -> [B, C, N] fp32 output.
    bn_relu_out_kernel<<<(unsigned)(((size_t)C_ * NCOL) / 256), 256, 0, stream>>>(
        Ybuf, bn_g1, bn_b1, mean1, rstd1, out);
}
